// MFCC_36069135352565
// MI455X (gfx1250) — compile-verified, matrix operand order checked by reading, not run on hardware
//
#include <hip/hip_runtime.h>
#include <cstdint>
#include <math.h>

#define SRATE 16000
#define WINN  400
#define HOPN  160
#define NFFT  512
#define NBIN  257       // NFFT/2+1
#define NMELS 40
#define NFRM  998
#define NBAT  64
#define TOTF  (NBAT * NFRM)   // 63872 frames, 63872/16 = 3992 tiles exactly
#define KPAD  416             // 400 padded to 13*32
#define NPAD  272             // 257 padded to 17*16
#define NTILES 17
#define KSTEPS 13
#define MELK  288             // 272 padded to 9*32 for mel WMMA
#define MELN  48              // 40 padded to 3*16
#define TWOPI 6.28318530717958647692f
#define DPI   3.14159265358979323846

typedef _Float16 v8h  __attribute__((ext_vector_type(8)));
typedef _Float16 v16h __attribute__((ext_vector_type(16)));
typedef float    v8f  __attribute__((ext_vector_type(8)));
typedef unsigned int u32x4 __attribute__((ext_vector_type(4)));
typedef int          i32x4 __attribute__((ext_vector_type(4)));
typedef int          i32x8 __attribute__((ext_vector_type(8)));

union H16 { v16h v; v8h h[2]; };

// ---------------------------------------------------------------------------
// Kernel 0: generate constants on device.
//   cosT/sinT: [NPAD][KPAD] f16 transposed DFT basis (zero outside n<257,k<400)
//   fbankT:    [MELN][MELK] f16 transposed mel filterbank (double math for bins)
//   dctm:      [40][40]  f32 DCT-II ortho
// ---------------------------------------------------------------------------
__global__ void mfcc_const_kernel(_Float16* __restrict__ cosT,
                                  _Float16* __restrict__ sinT,
                                  _Float16* __restrict__ fbankT,
                                  float* __restrict__ dctm) {
    int id = blockIdx.x * blockDim.x + threadIdx.x;

    if (id < NPAD * KPAD) {
        int n = id / KPAD, k = id % KPAD;
        _Float16 c = (_Float16)0.f, s = (_Float16)0.f;
        if (n < NBIN && k < WINN) {
            // exact mod-512 argument reduction keeps angles precise
            float ang = TWOPI * (float)((n * k) & (NFFT - 1)) / (float)NFFT;
            c = (_Float16)cosf(ang);
            s = (_Float16)sinf(ang);
        }
        cosT[id] = c;
        sinT[id] = s;
    }

    if (id < MELN * MELK) {
        int m = id / MELK, k = id % MELK;
        float v = 0.f;
        if (m < NMELS && k < NBIN) {
            double himel = 2595.0 * log10(1.0 + (SRATE * 0.5) / 700.0);
            auto binAt = [&](int i) {
                double mel = himel * (double)i / (double)(NMELS + 1);
                double hz  = 700.0 * (pow(10.0, mel / 2595.0) - 1.0);
                return floor((double)(NFFT + 1) * hz / (double)SRATE);
            };
            double blo = binAt(m), bc = binAt(m + 1), bhi = binAt(m + 2);
            if (k >= (int)blo && k < (int)bc)
                v = (float)(((double)k - blo) / (bc - blo));
            else if (k >= (int)bc && k < (int)bhi)
                v = (float)((bhi - (double)k) / (bhi - bc));
        }
        fbankT[id] = (_Float16)v;
    }

    if (id < NMELS * NMELS) {
        int kk = id / NMELS, n = id % NMELS;
        double c = cos(DPI * (double)kk * (2.0 * n + 1.0) / (2.0 * NMELS));
        double scale = (kk == 0) ? sqrt(1.0 / NMELS) : sqrt(2.0 / NMELS);
        dctm[id] = (float)(scale * c);
    }
}

// ---------------------------------------------------------------------------
// Kernel 1: pre-emphasis + framing + Hamming window -> f16 frames [TOTF][KPAD]
// ---------------------------------------------------------------------------
__global__ void mfcc_prep_kernel(const float* __restrict__ x,
                                 _Float16* __restrict__ frames) {
    int id = blockIdx.x * blockDim.x + threadIdx.x;
    if (id >= TOTF * KPAD) return;
    int f = id / KPAD, k = id % KPAD;
    _Float16 outv = (_Float16)0.f;
    if (k < WINN) {
        int b = f / NFRM, i = f % NFRM;
        int s = i * HOPN + k;
        const float* row = x + (size_t)b * 160000;
        float xv = row[s];
        float pe = (s == 0) ? xv : xv - 0.97f * row[s - 1];
        float w  = 0.54f - 0.46f * cosf(TWOPI * (float)k / (float)WINN);
        outv = (_Float16)(pe * w);
    }
    frames[id] = outv;
}

// ---------------------------------------------------------------------------
// Kernel 2: per 16-frame tile:
//   TDM tensor_load_to_lds (A tile) -> WMMA DFT (re+im) -> |X| (f16 in LDS)
//   -> WMMA mel projection -> log -> DCT -> deltas -> 39 features/frame.
//   4 wave32s per workgroup.
// ---------------------------------------------------------------------------
__global__ __launch_bounds__(128) void mfcc_main_kernel(
    const _Float16* __restrict__ frames,
    const _Float16* __restrict__ cosT,
    const _Float16* __restrict__ sinT,
    const _Float16* __restrict__ fbankT,
    const float* __restrict__ dctm,
    float* __restrict__ out) {

    __shared__ _Float16 Atile[16 * KPAD];     // 13312 B (TDM destination)
    __shared__ _Float16 magF16[16 * MELK];    //  9216 B
    __shared__ float melLds[16 * NMELS];      //  2560 B
    __shared__ float mfccLds[16 * NMELS];     //  2560 B
    __shared__ float d1Lds[16 * NMELS];       //  2560 B

    const int tid  = threadIdx.x;
    const int lane = tid & 31;
    const int wave = tid >> 5;
    const int l15  = lane & 15;
    const int kb   = (lane < 16) ? 0 : 8;     // ISA 16-bit A-matrix K base
    const int tile = blockIdx.x;

#if defined(__HIP_DEVICE_COMPILE__)
    // ---- Tensor Data Mover: DMA the 16x416 f16 tile (16 rows x 832B) to LDS.
    if (wave == 0) {
        typedef __attribute__((address_space(3))) _Float16 lds_f16;
        uint32_t lds_addr = (uint32_t)(uint64_t)(lds_f16*)Atile;
        uint64_t ga = (uint64_t)(uintptr_t)(frames + (size_t)tile * 16 * KPAD);

        u32x4 g0;
        g0[0] = 1u;                                   // count=1 valid descriptor
        g0[1] = lds_addr;                             // lds_addr[31:0]
        g0[2] = (uint32_t)ga;                         // global_addr[31:0]
        g0[3] = ((uint32_t)(ga >> 32) & 0x01FFFFFFu)  // global_addr[56:32]
              | (2u << 30);                           // type=2 (image)

        i32x8 g1;
        g1[0] = (int)(1u << 16);                      // data_size=2B, mask=0
        g1[1] = (int)((uint32_t)KPAD << 16);          // tensor_dim0=416
        g1[2] = (int)(16u << 16);                     // tensor_dim1=16
        g1[3] = (int)((uint32_t)KPAD << 16);          // tile_dim0=416
        g1[4] = 16;                                   // tile_dim1=16, tile_dim2=0
        g1[5] = KPAD;                                 // tensor_dim0_stride=416
        g1[6] = (int)((uint32_t)(KPAD * 16) << 16);   // tensor_dim1_stride lo
        g1[7] = 0;                                    // tensor_dim1_stride hi

        i32x4 gz = {0, 0, 0, 0};
#if __clang_major__ >= 23
        i32x8 gz8 = {0, 0, 0, 0, 0, 0, 0, 0};
        __builtin_amdgcn_tensor_load_to_lds(g0, g1, gz, gz, gz8, 0);
#else
        __builtin_amdgcn_tensor_load_to_lds(g0, g1, gz, gz, 0);
#endif
        __builtin_amdgcn_s_wait_tensorcnt(0);
    }
#endif

    // Zero the padded magnitude tile (cols 272..287 are never written by the
    // DFT stage but are consumed by the mel WMMA; WMMA propagates NaN).
    {
        uint32_t* mz = (uint32_t*)magF16;
        for (int j = tid; j < 16 * MELK / 2; j += 128) mz[j] = 0u;
    }
    __syncthreads();

    // ---- WMMA DFT: each wave owns N-tiles wave, wave+4, ... of 17 ----
    const _Float16* arow = Atile + l15 * KPAD;
    for (int nt = wave; nt < NTILES; nt += 4) {
        int ncol = nt * 16 + l15;
        const _Float16* brow_c = cosT + (size_t)ncol * KPAD;
        const _Float16* brow_s = sinT + (size_t)ncol * KPAD;
        __builtin_prefetch(brow_c, 0, 3);
        __builtin_prefetch(brow_s, 0, 3);

        v8f accR = {};
        v8f accI = {};
        for (int ks = 0; ks < KSTEPS; ++ks) {
            int ko = ks * 32 + kb;
            H16 a, bc, bs;
            a.h[0]  = *(const v8h*)(arow + ko);
            a.h[1]  = *(const v8h*)(arow + ko + 16);
            bc.h[0] = *(const v8h*)(brow_c + ko);
            bc.h[1] = *(const v8h*)(brow_c + ko + 16);
            bs.h[0] = *(const v8h*)(brow_s + ko);
            bs.h[1] = *(const v8h*)(brow_s + ko + 16);
            accR = __builtin_amdgcn_wmma_f32_16x16x32_f16(
                false, a.v, false, bc.v, (short)0, accR, false, false);
            accI = __builtin_amdgcn_wmma_f32_16x16x32_f16(
                false, a.v, false, bs.v, (short)0, accI, false, false);
        }
        // C/D layout: lanes 0-15 -> M=r, N=lane; lanes 16-31 -> M=r+8, N=lane-16
        #pragma unroll
        for (int r = 0; r < 8; ++r) {
            int M = (lane < 16) ? r : (r + 8);
            float re = accR[r], im = accI[r];
            magF16[M * MELK + nt * 16 + l15] = (_Float16)sqrtf(re * re + im * im);
        }
    }
    __syncthreads();

    // ---- mel projection as WMMA: mag[16x288] @ fbankT[288x48], waves 0..2 ----
    if (wave < 3) {
        int mcol = wave * 16 + l15;                 // mel index 0..47
        const _Float16* mrow = magF16 + l15 * MELK;
        const _Float16* brow = fbankT + (size_t)mcol * MELK;
        v8f accM = {};
        for (int ks = 0; ks < MELK / 32; ++ks) {
            int ko = ks * 32 + kb;
            H16 a, b;
            a.h[0] = *(const v8h*)(mrow + ko);
            a.h[1] = *(const v8h*)(mrow + ko + 16);
            b.h[0] = *(const v8h*)(brow + ko);
            b.h[1] = *(const v8h*)(brow + ko + 16);
            accM = __builtin_amdgcn_wmma_f32_16x16x32_f16(
                false, a.v, false, b.v, (short)0, accM, false, false);
        }
        #pragma unroll
        for (int r = 0; r < 8; ++r) {
            int M = (lane < 16) ? r : (r + 8);
            int col = wave * 16 + l15;
            if (col < NMELS)
                melLds[M * NMELS + col] = logf(accM[r] + 1e-20f);
        }
    }
    __syncthreads();

    // ---- DCT-II ortho (small: keep f32 VALU for precision) ----
    for (int p = tid; p < 16 * NMELS; p += 128) {
        int fr = p / NMELS, c = p % NMELS;
        const float* lrow = melLds + fr * NMELS;
        const float* drow = dctm + c * NMELS;
        float acc = 0.f;
        for (int n = 0; n < NMELS; ++n) acc += lrow[n] * drow[n];
        mfccLds[p] = acc;
    }
    __syncthreads();

    // ---- first delta along coefficient axis (zero at edges) ----
    for (int p = tid; p < 16 * NMELS; p += 128) {
        int fr = p / NMELS, c = p % NMELS;
        float v = 0.f;
        if (c >= 1 && c <= NMELS - 2)
            v = 0.5f * (mfccLds[fr * NMELS + c + 1] - mfccLds[fr * NMELS + c - 1]);
        d1Lds[p] = v;
    }
    __syncthreads();

    // ---- emit [mfcc[0:13], d1[0:13], d2[0:13]] per frame ----
    for (int p = tid; p < 16 * 39; p += 128) {
        int fr = p / 39, j = p % 39;
        float v;
        if (j < 13) {
            v = mfccLds[fr * NMELS + j];
        } else if (j < 26) {
            v = d1Lds[fr * NMELS + (j - 13)];
        } else {
            int c = j - 26;
            v = (c >= 1) ? 0.5f * (d1Lds[fr * NMELS + c + 1] - d1Lds[fr * NMELS + c - 1])
                         : 0.f;
        }
        out[(size_t)(tile * 16 + fr) * 39 + j] = v;
    }
}

// ---------------------------------------------------------------------------
extern "C" void kernel_launch(void* const* d_in, const int* in_sizes, int n_in,
                              void* d_out, int out_size, void* d_ws, size_t ws_size,
                              hipStream_t stream) {
    (void)in_sizes; (void)n_in; (void)out_size; (void)ws_size;
    const float* x = (const float*)d_in[0];
    float* out = (float*)d_out;

    char* ws = (char*)d_ws;
    size_t off = 0;
    auto take = [&](size_t bytes) {
        char* p = ws + off;
        off = (off + bytes + 255) & ~(size_t)255;
        return p;
    };
    _Float16* frames = (_Float16*)take((size_t)TOTF * KPAD * sizeof(_Float16)); // ~53 MB
    _Float16* cosT   = (_Float16*)take((size_t)NPAD * KPAD * sizeof(_Float16)); // 226 KB
    _Float16* sinT   = (_Float16*)take((size_t)NPAD * KPAD * sizeof(_Float16)); // 226 KB
    _Float16* fbankT = (_Float16*)take((size_t)MELN * MELK * sizeof(_Float16)); //  27 KB
    float*    dctm   = (float*)take((size_t)NMELS * NMELS * sizeof(float));     //   6 KB

    {   // constants
        int n = NPAD * KPAD;
        mfcc_const_kernel<<<(n + 255) / 256, 256, 0, stream>>>(cosT, sinT, fbankT, dctm);
    }
    {   // pre-emphasis + window + framing
        long long n = (long long)TOTF * KPAD;
        mfcc_prep_kernel<<<(unsigned)((n + 255) / 256), 256, 0, stream>>>(x, frames);
    }
    {   // fused TDM + WMMA DFT + WMMA mel + log/DCT/deltas
        mfcc_main_kernel<<<TOTF / 16, 128, 0, stream>>>(frames, cosT, sinT, fbankT, dctm, out);
    }
}
